// Block_45767171506179
// MI455X (gfx1250) — matrix operand order repaired and checked by reading, not compiled, not run on hardware
//
#include <hip/hip_runtime.h>
#include <hip/hip_bf16.h>
#include <math.h>

// ---------------- CDNA5 WMMA types & fragment helpers (wave32) ----------------
typedef __attribute__((ext_vector_type(16))) _Float16 v16h;
typedef __attribute__((ext_vector_type(8)))  _Float16 v8h;
typedef __attribute__((ext_vector_type(8)))  float    v8f;

#define NEMB 384
#define NHEAD 6
#define HS 64
#define TT 256
#define BB 128
#define FF 1536
#define NTOK (BB * TT)

__device__ __forceinline__ v8f wmma_f16(v16h a, v16h b, v8f c) {
  return __builtin_amdgcn_wmma_f32_16x16x32_f16(false, a, false, b, (short)0, c,
                                                false, false);
}

// A-fragment (M=16,K=32), row-major source. Lanes 0-15: M=lane, K={0..7,16..23};
// lanes 16-31: M=lane-16, K={8..15,24..31}.
__device__ __forceinline__ v16h load_a16(const _Float16* A, int lda, int row0, int k0) {
  int lane = threadIdx.x & 31;
  int l = lane & 15, hi = lane >> 4;
  const _Float16* p = A + (size_t)(row0 + l) * lda + k0 + hi * 8;
  v8h lo = *(const v8h*)p;
  v8h hh = *(const v8h*)(p + 16);
  v16h a;
#pragma unroll
  for (int e = 0; e < 8; ++e) { a[e] = lo[e]; a[8 + e] = hh[e]; }
  return a;
}

// B-fragment (K=32,N=16) where logical B[k][n] = M[n][k], M row-major [N x K]
// (i.e. weights stored pre-transposed): two 16-byte loads per lane.
// Lanes 0-15 hold K={0..7,16..23}, lanes 16-31 hold K={8..15,24..31}; N = lane&15.
__device__ __forceinline__ v16h load_bT16(const _Float16* M, int ldm, int k0, int col0) {
  int lane = threadIdx.x & 31;
  int n = lane & 15, hi = lane >> 4;
  const _Float16* p = M + (size_t)(col0 + n) * ldm + k0 + hi * 8;
  v8h lo = *(const v8h*)p;
  v8h hh = *(const v8h*)(p + 16);
  v16h b;
#pragma unroll
  for (int e = 0; e < 8; ++e) { b[e] = lo[e]; b[8 + e] = hh[e]; }
  return b;
}

__device__ __forceinline__ void store_c_h16(_Float16* D, int ldd, int row0, int col0, v8f c) {
  int lane = threadIdx.x & 31;
  int n = lane & 15, hi = lane >> 4;
#pragma unroll
  for (int r = 0; r < 8; ++r)
    D[(size_t)(row0 + hi * 8 + r) * ldd + col0 + n] = (_Float16)c[r];
}

// ---------------- f32 -> f16 convert with transpose (per-matrix batch) ----------------
// in: nmat row-major [R][C] fp32;  out: nmat row-major [C][R] f16
__global__ void cvt_t_f16(const float* __restrict__ in, _Float16* __restrict__ out,
                          int R, int C, int total) {
  int i = blockIdx.x * blockDim.x + threadIdx.x;
  if (i >= total) return;
  int rc = R * C;
  int m = i / rc;
  int rem = i - m * rc;
  int r = rem / C, c = rem - r * C;
  out[(size_t)m * rc + (size_t)c * R + r] = (_Float16)in[i];
}

// ---------------- layernorm: one wave per token ----------------
__global__ __launch_bounds__(256) void ln_kernel(const float* __restrict__ x,
                                                 const float* __restrict__ g,
                                                 const float* __restrict__ b,
                                                 _Float16* __restrict__ out) {
  int token = blockIdx.x * (blockDim.x >> 5) + (threadIdx.x >> 5);
  if (token >= NTOK) return;
  int lane = threadIdx.x & 31;
  const float* row = x + (size_t)token * NEMB;
  float s = 0.f, s2 = 0.f;
  for (int i = lane; i < NEMB; i += 32) { float v = row[i]; s += v; s2 += v * v; }
#pragma unroll
  for (int off = 16; off > 0; off >>= 1) {
    s  += __shfl_xor(s, off, 32);
    s2 += __shfl_xor(s2, off, 32);
  }
  float mu = s * (1.0f / NEMB);
  float var = s2 * (1.0f / NEMB) - mu * mu;
  float rstd = rsqrtf(var + 1e-5f);
  _Float16* orow = out + (size_t)token * NEMB;
  for (int i = lane; i < NEMB; i += 32)
    orow[i] = (_Float16)((row[i] - mu) * rstd * g[i] + b[i]);
}

// ---------------- QKV projection: one wave = one 16x64 strip of q/k/v ----------------
// Weights pre-transposed: wqT/wkT/wvT are [H][HS][NEMB] f16.
// q,k stored [B,H,T,HS]; v stored transposed [B,H,HS,T] for the P@V GEMM.
__global__ __launch_bounds__(256, 1) void qkv_kernel(const _Float16* __restrict__ h16,
                                                     const _Float16* __restrict__ wqT,
                                                     const _Float16* __restrict__ wkT,
                                                     const _Float16* __restrict__ wvT,
                                                     _Float16* __restrict__ q,
                                                     _Float16* __restrict__ k,
                                                     _Float16* __restrict__ v) {
  int wave = blockIdx.x * (blockDim.x >> 5) + (threadIdx.x >> 5);
  int mt = wave & 15; wave >>= 4;
  int hh = wave % NHEAD; wave /= NHEAD;
  int bb = wave & 127; wave >>= 7;
  int mat = wave;  // 0=q 1=k 2=v
  const _Float16* WT = (mat == 0 ? wqT : (mat == 1 ? wkT : wvT)) + (size_t)hh * HS * NEMB;
  const _Float16* A = h16 + (size_t)(bb * TT + mt * 16) * NEMB;
  v8f acc[4] = {v8f{}, v8f{}, v8f{}, v8f{}};
#pragma unroll
  for (int k0 = 0; k0 < NEMB; k0 += 32) {
    v16h a  = load_a16(A, NEMB, 0, k0);
    v16h b0 = load_bT16(WT, NEMB, k0, 0);
    v16h b1 = load_bT16(WT, NEMB, k0, 16);
    v16h b2 = load_bT16(WT, NEMB, k0, 32);
    v16h b3 = load_bT16(WT, NEMB, k0, 48);
    acc[0] = wmma_f16(a, b0, acc[0]);
    acc[1] = wmma_f16(a, b1, acc[1]);
    acc[2] = wmma_f16(a, b2, acc[2]);
    acc[3] = wmma_f16(a, b3, acc[3]);
  }
  int lane = threadIdx.x & 31;
  int n = lane & 15, hi = lane >> 4;
  if (mat == 2) {
    // transposed store: vT[(b,h)][d][t]
    _Float16* OT = v + (size_t)(bb * NHEAD + hh) * HS * TT;
#pragma unroll
    for (int j = 0; j < 4; ++j)
#pragma unroll
      for (int r = 0; r < 8; ++r)
        OT[(size_t)(j * 16 + n) * TT + mt * 16 + hi * 8 + r] = (_Float16)acc[j][r];
  } else {
    _Float16* O = (mat == 0 ? q : k) + ((size_t)(bb * NHEAD + hh) * TT + mt * 16) * HS;
#pragma unroll
    for (int j = 0; j < 4; ++j) store_c_h16(O, HS, 0, j * 16, acc[j]);
  }
}

// ---------------- fused causal attention: one block per (b,h), 4 waves ----------------
__global__ __launch_bounds__(128, 1) void attn_kernel(const _Float16* __restrict__ q,
                                                      const _Float16* __restrict__ k,
                                                      const _Float16* __restrict__ v,
                                                      _Float16* __restrict__ ao) {
  __shared__ __align__(32) _Float16 wbuf[4][16 * TT];  // 32 KB: per-wave P staging
  int bh = blockIdx.x;                 // b*NHEAD + h
  int wid = threadIdx.x >> 5;
  int lane = threadIdx.x & 31;
  int n = lane & 15, hi = lane >> 4;
  int bb = bh / NHEAD, hh = bh % NHEAD;
  const _Float16* Q  = q + (size_t)bh * TT * HS;
  const _Float16* K  = k + (size_t)bh * TT * HS;
  const _Float16* VT = v + (size_t)bh * HS * TT;       // [HS][TT]
  const float scale = 0.05103103630798288f;  // 384^-0.5 (full embed dim, per reference)
  _Float16* wl = &wbuf[wid][0];

  for (int mi = 0; mi < 4; ++mi) {
    int mt = mi * 4 + wid;             // interleave tile sizes across waves
    int t0 = mt * 16;
    int ntiles = mt + 1;               // causal: s-tiles 0..mt
    v16h qa0 = load_a16(Q, HS, t0, 0);
    v16h qa1 = load_a16(Q, HS, t0, 32);

    v8f sc[16];
    float rm[8], rs[8];
#pragma unroll
    for (int r = 0; r < 8; ++r) { rm[r] = -3.0e38f; rs[r] = 0.f; }

#pragma unroll
    for (int j = 0; j < 16; ++j) {
      if (j < ntiles) {
        v16h kb0 = load_bT16(K, HS, 0, j * 16);
        v16h kb1 = load_bT16(K, HS, 32, j * 16);
        v8f c = {};
        c = wmma_f16(qa0, kb0, c);
        c = wmma_f16(qa1, kb1, c);
        int s_col = j * 16 + n;
#pragma unroll
        for (int r = 0; r < 8; ++r) {
          int row = t0 + hi * 8 + r;
          c[r] = (s_col <= row) ? c[r] * scale : -3.0e38f;
          rm[r] = fmaxf(rm[r], c[r]);
        }
        sc[j] = c;
      }
    }
    // row max across the 16 lanes of each half (rows r / r+8 stay separated)
#pragma unroll
    for (int off = 8; off > 0; off >>= 1)
#pragma unroll
      for (int r = 0; r < 8; ++r) rm[r] = fmaxf(rm[r], __shfl_xor(rm[r], off, 32));
    // exp + row sum, stage P (f16) into LDS in row-major [16][256]
#pragma unroll
    for (int j = 0; j < 16; ++j) {
      if (j < ntiles) {
        v8f c = sc[j];
#pragma unroll
        for (int r = 0; r < 8; ++r) {
          float e = __expf(c[r] - rm[r]);
          rs[r] += e;
          wl[(hi * 8 + r) * TT + j * 16 + n] = (_Float16)e;
        }
      }
    }
#pragma unroll
    for (int off = 8; off > 0; off >>= 1)
#pragma unroll
      for (int r = 0; r < 8; ++r) rs[r] += __shfl_xor(rs[r], off, 32);
    // zero-pad odd tail so the K=32 WMMA steps see zeros
    if (ntiles & 1) {
#pragma unroll
      for (int r = 0; r < 8; ++r)
        wl[(hi * 8 + r) * TT + ntiles * 16 + n] = (_Float16)0.f;
    }
    // O = P @ V  (K over valid s range only); V is pre-transposed [HS][TT]
    int ksteps = (ntiles + 1) >> 1;
    v8f o[4] = {v8f{}, v8f{}, v8f{}, v8f{}};
    for (int ks = 0; ks < ksteps; ++ks) {
      v16h a  = load_a16(wl, TT, 0, ks * 32);
      v16h b0 = load_bT16(VT, TT, ks * 32, 0);
      v16h b1 = load_bT16(VT, TT, ks * 32, 16);
      v16h b2 = load_bT16(VT, TT, ks * 32, 32);
      v16h b3 = load_bT16(VT, TT, ks * 32, 48);
      o[0] = wmma_f16(a, b0, o[0]);
      o[1] = wmma_f16(a, b1, o[1]);
      o[2] = wmma_f16(a, b2, o[2]);
      o[3] = wmma_f16(a, b3, o[3]);
    }
    // normalize rows by 1/sum, write concat-head output f16 [NTOK][NEMB]
#pragma unroll
    for (int j = 0; j < 4; ++j) {
#pragma unroll
      for (int r = 0; r < 8; ++r) {
        int row = bb * TT + t0 + hi * 8 + r;
        int col = hh * HS + j * 16 + n;
        ao[(size_t)row * NEMB + col] = (_Float16)(o[j][r] / rs[r]);
      }
    }
  }
}

// ---------------- output projection + bias + residual -> x1 (fp32) ----------------
__global__ __launch_bounds__(256, 1) void proj_kernel(const _Float16* __restrict__ ao,
                                                      const _Float16* __restrict__ wpT,
                                                      const float* __restrict__ bp,
                                                      const float* __restrict__ x,
                                                      float* __restrict__ x1) {
  int wave = blockIdx.x * (blockDim.x >> 5) + (threadIdx.x >> 5);
  int nb = wave % (NEMB / 64);
  int rowt = wave / (NEMB / 64);
  int row0 = rowt * 16, col0 = nb * 64;
  int lane = threadIdx.x & 31;
  int n = lane & 15, hi = lane >> 4;
  const _Float16* A = ao + (size_t)row0 * NEMB;
  v8f acc[4] = {v8f{}, v8f{}, v8f{}, v8f{}};
#pragma unroll
  for (int k0 = 0; k0 < NEMB; k0 += 32) {
    v16h a  = load_a16(A, NEMB, 0, k0);
    v16h b0 = load_bT16(wpT, NEMB, k0, col0);
    v16h b1 = load_bT16(wpT, NEMB, k0, col0 + 16);
    v16h b2 = load_bT16(wpT, NEMB, k0, col0 + 32);
    v16h b3 = load_bT16(wpT, NEMB, k0, col0 + 48);
    acc[0] = wmma_f16(a, b0, acc[0]);
    acc[1] = wmma_f16(a, b1, acc[1]);
    acc[2] = wmma_f16(a, b2, acc[2]);
    acc[3] = wmma_f16(a, b3, acc[3]);
  }
#pragma unroll
  for (int j = 0; j < 4; ++j) {
#pragma unroll
    for (int r = 0; r < 8; ++r) {
      int rr = row0 + hi * 8 + r;
      int cc = col0 + j * 16 + n;
      x1[(size_t)rr * NEMB + cc] = acc[j][r] + bp[cc] + x[(size_t)rr * NEMB + cc];
    }
  }
}

// ---------------- fused FFN: relu(h2@W1+b1)@W2 + b2 + x1 -> out ----------------
// w1T: [FF][NEMB] f16 (W1 transposed), w2T: [NEMB][FF] f16 (W2 transposed)
__global__ __launch_bounds__(256, 1) void ffn_kernel(const _Float16* __restrict__ h2,
                                                     const _Float16* __restrict__ w1T,
                                                     const float* __restrict__ b1,
                                                     const _Float16* __restrict__ w2T,
                                                     const float* __restrict__ b2,
                                                     const float* __restrict__ x1,
                                                     float* __restrict__ out) {
  __shared__ __align__(32) _Float16 sH[16 * NEMB];  // 12 KB activation tile
  __shared__ __align__(32) _Float16 sT[16 * FF];    // 48 KB hidden tile
  int row0 = blockIdx.x * 16;
  int wid = threadIdx.x >> 5;
  int lane = threadIdx.x & 31;
  int n = lane & 15, hi = lane >> 4;

  {  // cooperative load of 16xC activation tile
    const uint32_t* src = (const uint32_t*)(h2 + (size_t)row0 * NEMB);
    uint32_t* dst = (uint32_t*)sH;
    for (int i = threadIdx.x; i < 16 * NEMB / 2; i += 256) dst[i] = src[i];
  }
  __syncthreads();

  // phase 1: hidden = relu(sH @ W1 + b1); each wave owns 12 of 96 N-tiles.
  // Loop-interchanged: one LDS A-fragment load per k-step feeds 12 WMMAs.
  {
    v8f acc[12];
#pragma unroll
    for (int j = 0; j < 12; ++j) acc[j] = v8f{};
    int colbase = wid * 12 * 16;
    for (int k0 = 0; k0 < NEMB; k0 += 32) {
      v16h a = load_a16(sH, NEMB, 0, k0);
      __builtin_prefetch(w1T + (size_t)colbase * NEMB + k0 + 32, 0, 1);
#pragma unroll
      for (int j = 0; j < 12; ++j)
        acc[j] = wmma_f16(a, load_bT16(w1T, NEMB, k0, colbase + j * 16), acc[j]);
    }
#pragma unroll
    for (int j = 0; j < 12; ++j) {
#pragma unroll
      for (int r = 0; r < 8; ++r) {
        int cc = colbase + j * 16 + n;
        float val = acc[j][r] + b1[cc];
        val = val > 0.f ? val : 0.f;
        sT[(hi * 8 + r) * FF + cc] = (_Float16)val;
      }
    }
  }
  __syncthreads();

  // phase 2: out = sT @ W2 + b2 + x1; each wave owns 3 of 24 N-tiles.
  // Software-pipelined: next k-step's fragments preloaded before current WMMAs.
  {
    v8f acc[3] = {v8f{}, v8f{}, v8f{}};
    int colbase = wid * 3 * 16;
    v16h a   = load_a16(sT, FF, 0, 0);
    v16h b0  = load_bT16(w2T, FF, 0, colbase);
    v16h b1f = load_bT16(w2T, FF, 0, colbase + 16);
    v16h b2f = load_bT16(w2T, FF, 0, colbase + 32);
    for (int k0 = 0; k0 < FF - 32; k0 += 32) {
      int kn = k0 + 32;
      v16h an   = load_a16(sT, FF, 0, kn);
      v16h bn0  = load_bT16(w2T, FF, kn, colbase);
      v16h bn1  = load_bT16(w2T, FF, kn, colbase + 16);
      v16h bn2  = load_bT16(w2T, FF, kn, colbase + 32);
      __builtin_prefetch(w2T + (size_t)colbase * FF + kn + 32, 0, 1);
      acc[0] = wmma_f16(a, b0, acc[0]);
      acc[1] = wmma_f16(a, b1f, acc[1]);
      acc[2] = wmma_f16(a, b2f, acc[2]);
      a = an; b0 = bn0; b1f = bn1; b2f = bn2;
    }
    acc[0] = wmma_f16(a, b0, acc[0]);
    acc[1] = wmma_f16(a, b1f, acc[1]);
    acc[2] = wmma_f16(a, b2f, acc[2]);
#pragma unroll
    for (int j = 0; j < 3; ++j) {
#pragma unroll
      for (int r = 0; r < 8; ++r) {
        int rr = row0 + hi * 8 + r;
        int cc = colbase + j * 16 + n;
        out[(size_t)rr * NEMB + cc] = acc[j][r] + b2[cc] + x1[(size_t)rr * NEMB + cc];
      }
    }
  }
}

// ---------------- host-side orchestration ----------------
extern "C" void kernel_launch(void* const* d_in, const int* in_sizes, int n_in,
                              void* d_out, int out_size, void* d_ws, size_t ws_size,
                              hipStream_t stream) {
  const float* x   = (const float*)d_in[0];
  const float* Wq  = (const float*)d_in[1];
  const float* Wk  = (const float*)d_in[2];
  const float* Wv  = (const float*)d_in[3];
  const float* Wp  = (const float*)d_in[4];
  const float* bp  = (const float*)d_in[5];
  const float* W1  = (const float*)d_in[6];
  const float* b1  = (const float*)d_in[7];
  const float* W2  = (const float*)d_in[8];
  const float* b2  = (const float*)d_in[9];
  const float* g1  = (const float*)d_in[10];
  const float* be1 = (const float*)d_in[11];
  const float* g2  = (const float*)d_in[12];
  const float* be2 = (const float*)d_in[13];
  float* outp = (float*)d_out;

  char* ws = (char*)d_ws;
  size_t off = 0;
  auto ALLOC = [&](size_t bytes) -> char* {
    char* p = ws + off;
    off += (bytes + 255) & ~(size_t)255;
    return p;
  };
  _Float16* h16  = (_Float16*)ALLOC((size_t)NTOK * NEMB * 2);
  _Float16* q16  = (_Float16*)ALLOC((size_t)BB * NHEAD * TT * HS * 2);
  _Float16* k16  = (_Float16*)ALLOC((size_t)BB * NHEAD * TT * HS * 2);
  _Float16* v16t = (_Float16*)ALLOC((size_t)BB * NHEAD * HS * TT * 2);
  _Float16* ao16 = (_Float16*)ALLOC((size_t)NTOK * NEMB * 2);
  float*    x1f  = (float*)   ALLOC((size_t)NTOK * NEMB * 4);
  _Float16* h216 = (_Float16*)ALLOC((size_t)NTOK * NEMB * 2);
  _Float16* wqT  = (_Float16*)ALLOC((size_t)NHEAD * NEMB * HS * 2);
  _Float16* wkT  = (_Float16*)ALLOC((size_t)NHEAD * NEMB * HS * 2);
  _Float16* wvT  = (_Float16*)ALLOC((size_t)NHEAD * NEMB * HS * 2);
  _Float16* wpT  = (_Float16*)ALLOC((size_t)NEMB * NEMB * 2);
  _Float16* w1T  = (_Float16*)ALLOC((size_t)NEMB * FF * 2);
  _Float16* w2T  = (_Float16*)ALLOC((size_t)FF * NEMB * 2);

  auto cvtT = [&](const float* src, _Float16* dst, int R, int C, int nmat) {
    int total = nmat * R * C;
    cvt_t_f16<<<(total + 255) / 256, 256, 0, stream>>>(src, dst, R, C, total);
  };
  cvtT(Wq, wqT, NEMB, HS, NHEAD);   // -> [H][HS][NEMB]
  cvtT(Wk, wkT, NEMB, HS, NHEAD);
  cvtT(Wv, wvT, NEMB, HS, NHEAD);
  cvtT(Wp, wpT, NEMB, NEMB, 1);     // -> [N][K]
  cvtT(W1, w1T, NEMB, FF, 1);       // -> [FF][NEMB]
  cvtT(W2, w2T, FF, NEMB, 1);       // -> [NEMB][FF]

  ln_kernel<<<NTOK / 8, 256, 0, stream>>>(x, g1, be1, h16);

  // 3 mats * B * H * (T/16) waves, 8 waves per block
  qkv_kernel<<<(3 * BB * NHEAD * (TT / 16)) / 8, 256, 0, stream>>>(
      h16, wqT, wkT, wvT, q16, k16, v16t);

  attn_kernel<<<BB * NHEAD, 128, 0, stream>>>(q16, k16, v16t, ao16);

  proj_kernel<<<((NTOK / 16) * (NEMB / 64)) / 8, 256, 0, stream>>>(
      ao16, wpT, bp, x, x1f);

  ln_kernel<<<NTOK / 8, 256, 0, stream>>>(x1f, g2, be2, h216);

  ffn_kernel<<<NTOK / 16, 256, 0, stream>>>(h216, w1T, b1, w2T, b2, x1f, outp);
}
